// AtomBondGraph_61984968015930
// MI455X (gfx1250) — matrix operand order repaired and checked by reading, not compiled, not run on hardware
//
#include <hip/hip_runtime.h>

// AtomBondGraph GNN (GIN) for MI455X / gfx1250, wave32 + WMMA bf16 + async LDS copy.
// B=256, N=256, D=256, E=262144, L=5, VOCAB=16, V=65536.

typedef __attribute__((ext_vector_type(16))) __bf16 v16bf;
typedef __attribute__((ext_vector_type(2)))  __bf16 v2bf;
typedef __attribute__((ext_vector_type(8)))  float  v8f;

#define DD   256
#define BB   256
#define NN   256
#define EE   262144
#define LL   5
#define VV   (BB * NN)          // 65536
#define EV   (EE + VV)          // 327680
#define VOC  16

union Frag { v16bf v; uint4 u[2]; };

__device__ __forceinline__ unsigned pack2bf(float lo, float hi) {
#if __has_builtin(__builtin_amdgcn_cvt_pk_bf16_f32)
    v2bf r = __builtin_amdgcn_cvt_pk_bf16_f32(lo, hi);
    return __builtin_bit_cast(unsigned, r);
#else
    union { __bf16 b; unsigned short u; } a, b;
    a.b = (__bf16)lo; b.b = (__bf16)hi;
    return (unsigned)a.u | ((unsigned)b.u << 16);
#endif
}

// ---------------------------------------------------------------- utilities

__global__ __launch_bounds__(256) void zero4_kernel(float4* __restrict__ p, int n4) {
    int i = blockIdx.x * 256 + threadIdx.x;
    if (i < n4) p[i] = make_float4(0.f, 0.f, 0.f, 0.f);
}

// convert fp32 [L][K][N] -> bf16 transposed [L][N][K]
__global__ __launch_bounds__(256) void cvt_tr_bf16_kernel(const float* __restrict__ src,
                                                          __bf16* __restrict__ dst,
                                                          int K, int N) {
    int i = blockIdx.x * 256 + threadIdx.x;       // over L*K*N, output-major
    int total = LL * K * N;
    if (i >= total) return;
    int layer = i / (K * N);
    int r     = i % (K * N);
    int n = r / K, k = r % K;
    dst[i] = (__bf16)src[(size_t)layer * K * N + (size_t)k * N + n];
}

// h = x[:, 1:, :] flattened to [V, D]   (float4 over V*D/4)
__global__ __launch_bounds__(256) void init_h_kernel(const float4* __restrict__ x,
                                                     float4* __restrict__ h) {
    int i = blockIdx.x * 256 + threadIdx.x;       // V*D/4
    int d4   = i & 63;                            // D/4 = 64
    int node = i >> 6;
    int b    = node >> 8;
    int n    = node & (NN - 1);
    h[i] = x[((size_t)b * (NN + 1) + (n + 1)) * 64 + d4];
}

// ------------------------------------------------------------- edge phase
// One wave per edge; lane covers 2 float4s of the 256-wide feature row.
__global__ __launch_bounds__(256) void edge_kernel(
    const float* __restrict__ h, const float* __restrict__ ew,
    const int* __restrict__ eidx, const int* __restrict__ eattr,
    const float* __restrict__ bemb /* layer base: [3][VOC][D] */,
    float* __restrict__ aggr) {
    int e    = blockIdx.x * 8 + (threadIdx.x >> 5);
    int lane = threadIdx.x & 31;
    if (e >= EV) return;
    int s, dn, a0, a1, a2; float w;
    if (e < EE) {
        s  = eidx[e];       dn = eidx[EE + e];
        a0 = eattr[e * 3];  a1 = eattr[e * 3 + 1];  a2 = eattr[e * 3 + 2];
        w  = ew[e];
    } else {                         // self loops: attr = [5, 7, 0], w = 1
        int nd = e - EE; s = nd; dn = nd; a0 = 5; a1 = 7; a2 = 0; w = 1.0f;
    }
    const float4* hs = (const float4*)(h + (size_t)s * DD);
    const float4* e0 = (const float4*)(bemb + (size_t)a0 * DD);
    const float4* e1 = (const float4*)(bemb + (size_t)(VOC + a1) * DD);
    const float4* e2 = (const float4*)(bemb + (size_t)(2 * VOC + a2) * DD);
    float*        ad = aggr + (size_t)dn * DD;
    #pragma unroll
    for (int j = 0; j < 2; j++) {
        int d4 = j * 32 + lane;                   // float4 index 0..63
        float4 hv = hs[d4], v0 = e0[d4], v1 = e1[d4], v2 = e2[d4];
        float m0 = hv.x + v0.x + v1.x + v2.x;
        float m1 = hv.y + v0.y + v1.y + v2.y;
        float m2 = hv.z + v0.z + v1.z + v2.z;
        float m3 = hv.w + v0.w + v1.w + v2.w;
        m0 = (m0 > 0.f) ? m0 * w : 0.f;
        m1 = (m1 > 0.f) ? m1 * w : 0.f;
        m2 = (m2 > 0.f) ? m2 * w : 0.f;
        m3 = (m3 > 0.f) ? m3 * w : 0.f;
        unsafeAtomicAdd(&ad[d4 * 4 + 0], m0);
        unsafeAtomicAdd(&ad[d4 * 4 + 1], m1);
        unsafeAtomicAdd(&ad[d4 * 4 + 2], m2);
        unsafeAtomicAdd(&ad[d4 * 4 + 3], m3);
    }
}

// ------------------------------------------------------------- GEMM + BN stats
// C[M,N] = A[M,K] @ B[K,N] + bias, fused per-column sum/sumsq for BatchNorm.
// B supplied pre-transposed bf16: BwT[N][K]. N, K compile-time for address math.
// MODE 0: A = (1+eps)*A0 + A1              (h, aggr)
// MODE 1: A = relu(A0*scale[k] + shift[k]) (bn1-applied z1)
template <int MODE, int N, int K>
__global__ __launch_bounds__(256) void gemm_bn_kernel(
    const float* __restrict__ A0, const float* __restrict__ A1,
    const float* __restrict__ epsp,
    const float* __restrict__ scale, const float* __restrict__ shift,
    const __bf16* __restrict__ BwT, const float* __restrict__ bias,
    float* __restrict__ C, float* __restrict__ sums, float* __restrict__ sumsq) {
    constexpr int SA = 40, SB = 40;               // bf16 elems; keeps 16B alignment
    __shared__ __align__(16) __bf16 As[128 * SA];
    __shared__ __align__(16) __bf16 Bs[128 * SB];

    const int tid  = threadIdx.x;
    const int lane = tid & 31;
    const int wave = tid >> 5;
    const int wm   = wave >> 2;                    // 0..1  (64 rows each)
    const int wn   = wave & 3;                     // 0..3  (32 cols each)
    const int half = lane >> 4;
    const int l16  = lane & 15;

    const int m0 = blockIdx.y * 128;
    const int n0 = blockIdx.x * 128;

    const int arow = tid >> 1;                     // 0..127
    const int ac0  = (tid & 1) * 16;               // 0 / 16
    const int brow = tid >> 1;                     // B tile row (n) 0..127
    const int bc0  = (tid & 1) * 16;               // k-chunk 0 / 16

    const float ep = (MODE == 0) ? (1.0f + epsp[0]) : 0.0f;

    const float4* pA0 = (const float4*)(A0 + (size_t)(m0 + arow) * K);
    const float4* pA1 = (MODE == 0) ? (const float4*)(A1 + (size_t)(m0 + arow) * K)
                                    : nullptr;
    const float4* psc = (MODE == 1) ? (const float4*)scale : nullptr;
    const float4* psh = (MODE == 1) ? (const float4*)shift : nullptr;

    const unsigned lbs = (unsigned)(uintptr_t)&Bs[brow * SB + bc0];
    const __bf16*  gb  = BwT + (size_t)(n0 + brow) * K + bc0;

    v8f zacc = {0.f, 0.f, 0.f, 0.f, 0.f, 0.f, 0.f, 0.f};
    v8f acc[4][2];
    #pragma unroll
    for (int i = 0; i < 4; i++) { acc[i][0] = zacc; acc[i][1] = zacc; }

    for (int k0 = 0; k0 < K; k0 += 32) {
        __syncthreads();

        // B tile: straight async row copy of pre-transposed weights -> Bs[n][k]
        {
            unsigned long long ga = (unsigned long long)(gb + k0);
            asm volatile("global_load_async_to_lds_b128 %0, %1, off"
                         :: "v"(lbs), "v"(ga) : "memory");
            asm volatile("global_load_async_to_lds_b128 %0, %1, off offset:16"
                         :: "v"(lbs), "v"(ga) : "memory");
        }

        // A tile: 128x32 fp32 -> bf16 with fused prologue, packed b128 stores
        {
            const int kb = (k0 + ac0) >> 2;        // float4 index of first chunk
            #pragma unroll
            for (int g = 0; g < 2; g++) {          // 2 x 8 floats
                float4 u, v;
                if (MODE == 0) {
                    float4 a = pA0[kb + 2 * g],     b = pA0[kb + 2 * g + 1];
                    float4 c = pA1[kb + 2 * g],     d = pA1[kb + 2 * g + 1];
                    u = make_float4(ep * a.x + c.x, ep * a.y + c.y,
                                    ep * a.z + c.z, ep * a.w + c.w);
                    v = make_float4(ep * b.x + d.x, ep * b.y + d.y,
                                    ep * b.z + d.z, ep * b.w + d.w);
                } else {
                    float4 a = pA0[kb + 2 * g],     b = pA0[kb + 2 * g + 1];
                    float4 s0 = psc[kb + 2 * g],    s1 = psc[kb + 2 * g + 1];
                    float4 t0 = psh[kb + 2 * g],    t1 = psh[kb + 2 * g + 1];
                    u = make_float4(fmaxf(a.x * s0.x + t0.x, 0.f),
                                    fmaxf(a.y * s0.y + t0.y, 0.f),
                                    fmaxf(a.z * s0.z + t0.z, 0.f),
                                    fmaxf(a.w * s0.w + t0.w, 0.f));
                    v = make_float4(fmaxf(b.x * s1.x + t1.x, 0.f),
                                    fmaxf(b.y * s1.y + t1.y, 0.f),
                                    fmaxf(b.z * s1.z + t1.z, 0.f),
                                    fmaxf(b.w * s1.w + t1.w, 0.f));
                }
                uint4 P;
                P.x = pack2bf(u.x, u.y); P.y = pack2bf(u.z, u.w);
                P.z = pack2bf(v.x, v.y); P.w = pack2bf(v.z, v.w);
                *(uint4*)&As[arow * SA + ac0 + g * 8] = P;
            }
        }

        asm volatile("s_wait_asynccnt 0x0" ::: "memory");
        __syncthreads();

        Frag a[4], b[2];
        #pragma unroll
        for (int ms = 0; ms < 4; ms++) {           // A frag: K 0-7 / 16-23 per half
            const int r = wm * 64 + ms * 16 + l16;
            const __bf16* p = &As[r * SA + half * 8];
            a[ms].u[0] = *(const uint4*)p;
            a[ms].u[1] = *(const uint4*)(p + 16);
        }
        #pragma unroll
        for (int ns = 0; ns < 2; ns++) {           // B frag: K 0-7 / 16-23 per half
            const int c = wn * 32 + ns * 16 + l16;
            const __bf16* p = &Bs[c * SB + half * 8];
            b[ns].u[0] = *(const uint4*)p;
            b[ns].u[1] = *(const uint4*)(p + 16);
        }
        #pragma unroll
        for (int ms = 0; ms < 4; ms++)
            #pragma unroll
            for (int ns = 0; ns < 2; ns++)
                acc[ms][ns] = __builtin_amdgcn_wmma_f32_16x16x32_bf16(
                    false, a[ms].v, false, b[ns].v, (short)0, acc[ms][ns],
                    false, false);
    }

    // epilogue: + bias, store fp32, fused column sum/sumsq for BatchNorm
    float s[2] = {0.f, 0.f}, q[2] = {0.f, 0.f};
    #pragma unroll
    for (int ms = 0; ms < 4; ms++) {
        #pragma unroll
        for (int ns = 0; ns < 2; ns++) {
            const int col  = n0 + wn * 32 + ns * 16 + l16;
            const float bi = bias[col];
            const int rowb = m0 + wm * 64 + ms * 16 + half * 8;
            float* Cp = C + (size_t)rowb * N + col;
            #pragma unroll
            for (int r = 0; r < 8; r++) {
                float z = acc[ms][ns][r] + bi;
                Cp[(size_t)r * N] = z;
                s[ns] += z; q[ns] += z * z;
            }
        }
    }
    #pragma unroll
    for (int ns = 0; ns < 2; ns++) {               // lanes l and l+16 share a column
        float ss = s[ns] + __shfl_xor(s[ns], 16, 32);
        float qq = q[ns] + __shfl_xor(q[ns], 16, 32);
        if (half == 0) {
            const int col = n0 + wn * 32 + ns * 16 + l16;
            unsafeAtomicAdd(&sums[col],  ss);
            unsafeAtomicAdd(&sumsq[col], qq);
        }
    }
}

// stats -> fused affine: scale = g*rsqrt(var+eps), shift = b - mean*scale
__global__ __launch_bounds__(256) void bn_finalize_kernel(
    const float* __restrict__ s, const float* __restrict__ q,
    const float* __restrict__ gam, const float* __restrict__ bet,
    float* __restrict__ scale, float* __restrict__ shift, int n, float cnt) {
    int i = blockIdx.x * 256 + threadIdx.x;
    if (i < n) {
        float m  = s[i] / cnt;
        float v  = q[i] / cnt - m * m;
        float r  = rsqrtf(v + 1e-5f);
        float sc = r * gam[i];
        scale[i] = sc;
        shift[i] = bet[i] - m * sc;
    }
}

__global__ __launch_bounds__(256) void bn_apply_kernel(
    const float4* __restrict__ z, const float4* __restrict__ scale,
    const float4* __restrict__ shift, float4* __restrict__ h, int n4, int relu) {
    int i = blockIdx.x * 256 + threadIdx.x;       // over V*D/4
    if (i >= n4) return;
    int c4 = i & 63;                              // D/4 = 64
    float4 a = z[i], sc = scale[c4], sh = shift[c4];
    float4 v = make_float4(a.x * sc.x + sh.x, a.y * sc.y + sh.y,
                           a.z * sc.z + sh.z, a.w * sc.w + sh.w);
    if (relu) {
        v.x = fmaxf(v.x, 0.f); v.y = fmaxf(v.y, 0.f);
        v.z = fmaxf(v.z, 0.f); v.w = fmaxf(v.w, 0.f);
    }
    h[i] = v;
}

// out = x; out[:, 1:, :] += h
__global__ __launch_bounds__(256) void out_kernel(
    const float4* __restrict__ x, const float4* __restrict__ h,
    float4* __restrict__ out, int n4) {
    int i = blockIdx.x * 256 + threadIdx.x;
    if (i >= n4) return;
    int d4   = i & 63;
    int rest = i >> 6;
    int t    = rest % (NN + 1);
    int b    = rest / (NN + 1);
    float4 v = x[i];
    if (t > 0) {
        float4 r = h[((size_t)(b * NN + (t - 1))) * 64 + d4];
        v.x += r.x; v.y += r.y; v.z += r.z; v.w += r.w;
    }
    out[i] = v;
}

// ---------------------------------------------------------------- launcher

extern "C" void kernel_launch(void* const* d_in, const int* in_sizes, int n_in,
                              void* d_out, int out_size, void* d_ws, size_t ws_size,
                              hipStream_t stream) {
    const float* x    = (const float*)d_in[0];
    const float* ew   = (const float*)d_in[1];
    const float* bemb = (const float*)d_in[2];
    const float* W1   = (const float*)d_in[3];
    const float* b1   = (const float*)d_in[4];
    const float* g1   = (const float*)d_in[5];
    const float* bb1  = (const float*)d_in[6];
    const float* W2   = (const float*)d_in[7];
    const float* b2   = (const float*)d_in[8];
    const float* g2   = (const float*)d_in[9];
    const float* bb2  = (const float*)d_in[10];
    const float* eps  = (const float*)d_in[11];
    // d_in[12] = x_mask (all ones, unused)
    const int*   eidx  = (const int*)d_in[13];
    const int*   eattr = (const int*)d_in[14];

    // workspace layout
    char* w = (char*)d_ws;
    float*  h    = (float*)(w);                               // V*D   = 64 MB
    float*  aggr = (float*)(w + 67108864);                    // V*D   = 64 MB (reused as z2)
    float*  z1   = (float*)(w + 134217728);                   // V*2D  = 128 MB
    __bf16* W1bT = (__bf16*)(w + 268435456);                  // [L][512][256] bf16
    __bf16* W2bT = (__bf16*)(w + 269746176);                  // [L][256][512] bf16
    float*  st   = (float*)(w + 271056896);                   // stats (16 KB)
    float* sums1 = st,        *sumsq1 = st + 512;
    float* scl1  = st + 1024, *shf1   = st + 1536;
    float* sums2 = st + 2048, *sumsq2 = st + 2304;
    float* scl2  = st + 2560, *shf2   = st + 2816;

    const int VD  = VV * DD;                                  // 16,777,216
    const int WEL = LL * DD * 2 * DD;                         // 655,360 per weight array

    init_h_kernel<<<VD / 4 / 256, 256, 0, stream>>>((const float4*)x, (float4*)h);
    cvt_tr_bf16_kernel<<<(WEL + 255) / 256, 256, 0, stream>>>(W1, W1bT, DD, 2 * DD);
    cvt_tr_bf16_kernel<<<(WEL + 255) / 256, 256, 0, stream>>>(W2, W2bT, 2 * DD, DD);

    for (int i = 0; i < LL; i++) {
        zero4_kernel<<<VD / 4 / 256, 256, 0, stream>>>((float4*)aggr, VD / 4);
        zero4_kernel<<<4, 256, 0, stream>>>((float4*)st, 1024);

        edge_kernel<<<EV / 8, 256, 0, stream>>>(
            h, ew, eidx, eattr, bemb + (size_t)i * 3 * VOC * DD, aggr);

        // z1 = ((1+eps)h + aggr) @ W1 + b1   [M=V, N=512, K=256]
        gemm_bn_kernel<0, 512, 256><<<dim3(4, VV / 128), 256, 0, stream>>>(
            h, aggr, eps + i, nullptr, nullptr,
            W1bT + (size_t)i * DD * 2 * DD, b1 + (size_t)i * 2 * DD,
            z1, sums1, sumsq1);
        bn_finalize_kernel<<<2, 256, 0, stream>>>(
            sums1, sumsq1, g1 + (size_t)i * 2 * DD, bb1 + (size_t)i * 2 * DD,
            scl1, shf1, 2 * DD, (float)VV);

        // z2 = relu(bn1(z1)) @ W2 + b2       [M=V, N=256, K=512] -> aggr buffer
        gemm_bn_kernel<1, 256, 512><<<dim3(2, VV / 128), 256, 0, stream>>>(
            z1, nullptr, nullptr, scl1, shf1,
            W2bT + (size_t)i * DD * 2 * DD, b2 + (size_t)i * DD,
            aggr, sums2, sumsq2);
        bn_finalize_kernel<<<1, 256, 0, stream>>>(
            sums2, sumsq2, g2 + (size_t)i * DD, bb2 + (size_t)i * DD,
            scl2, shf2, DD, (float)VV);

        // h = maybe_relu(bn2(z2))
        bn_apply_kernel<<<VD / 4 / 256, 256, 0, stream>>>(
            (const float4*)aggr, (const float4*)scl2, (const float4*)shf2,
            (float4*)h, VD / 4, (i != LL - 1) ? 1 : 0);
    }

    const int TOT4 = BB * (NN + 1) * DD / 4;                  // 4,210,688
    out_kernel<<<(TOT4 + 255) / 256, 256, 0, stream>>>(
        (const float4*)x, (const float4*)h, (float4*)d_out, TOT4);
}
